// GCNConv_dgl_2817498546212
// MI455X (gfx1250) — matrix operand order repaired and checked by reading, not compiled, not run on hardware
//
#include <hip/hip_runtime.h>

#define N_NODES 50000
#define N_EDGES 800000
#define IN_DIM  512
#define OUT_DIM 256

typedef __attribute__((ext_vector_type(16))) __bf16 v16bf;
typedef __attribute__((ext_vector_type(8)))  __bf16 v8bf;
typedef __attribute__((ext_vector_type(4)))  __bf16 v4bf;
typedef __attribute__((ext_vector_type(2)))  __bf16 v2bf;
typedef __attribute__((ext_vector_type(8)))  float  v8f;
typedef __attribute__((ext_vector_type(4)))  float  v4f;

#define BM 64
#define BN 128
#define BK 32
#define LDA 40   // padded bf16 row stride of A tile (breaks LDS bank conflicts)
#define LDB 40   // padded bf16 column stride of B tile (stored column-major)

// ---------------------------------------------------------------------------
// Kernel 1: h = x @ W + b   (bf16 WMMA, f32 accumulate)
// grid = (ceil(M/64), 256/128), block = 256 threads = 8 wave32
// ---------------------------------------------------------------------------
__global__ __launch_bounds__(256) void gemm_bias_kernel(
    const float* __restrict__ x, const float* __restrict__ W,
    const float* __restrict__ bias, float* __restrict__ h)
{
    __shared__ __align__(16) __bf16 Alds[BM * LDA];   // [row][k] row-major
    __shared__ __align__(16) __bf16 Blds[BN * LDB];   // [col][k] column-major

    const int t     = threadIdx.x;
    const int mBase = blockIdx.x * BM;
    const int nBase = blockIdx.y * BN;

    const int wave = t >> 5;
    const int lane = t & 31;
    const int half = lane >> 4;   // 0: lanes 0-15, 1: lanes 16-31
    const int l16  = lane & 15;
    const int wm   = wave & 3;    // 16-row M tile within the 64-row block
    const int wn   = wave >> 2;   // 64-col N half within the 128-col block

    v8f acc[4] = {};              // 4 N-tiles of 16x16 f32

    for (int kBase = 0; kBase < IN_DIM; kBase += BK) {
        // ---- stage A tile (64x32 f32 -> bf16), packed ds_store_b64 ----
        #pragma unroll
        for (int i = 0; i < 2; i++) {
            int linear = t + i * 256;          // float4 units; 512 total
            int row = linear >> 3;             // 8 float4 per 32-wide row
            int cv  = linear & 7;
            int grow = mBase + row;
            if (grow >= N_NODES) grow = N_NODES - 1;   // clamp (stores guarded)
            v4f v = *(const v4f*)(x + (size_t)grow * IN_DIM + kBase + cv * 4);
            v4bf p = { (__bf16)v.x, (__bf16)v.y, (__bf16)v.z, (__bf16)v.w };
            *(v4bf*)&Alds[row * LDA + cv * 4] = p;     // one ds_store_b64
        }
        // ---- stage B tile (32x128 f32 -> bf16 transposed): each thread
        //      loads the same float4 n-chunk from two adjacent K rows, so the
        //      transposed write is a packed-pair ds_store_b32 along K ----
        #pragma unroll
        for (int i = 0; i < 2; i++) {
            int linear = t + i * 256;          // 512 work items
            int kp = linear >> 5;              // K pair index 0..15
            int cv = linear & 31;              // float4 column chunk
            const float* wp = W + (size_t)(kBase + kp * 2) * OUT_DIM + nBase + cv * 4;
            v4f v0 = *(const v4f*)(wp);
            v4f v1 = *(const v4f*)(wp + OUT_DIM);
            #pragma unroll
            for (int j = 0; j < 4; j++) {
                v2bf p = { (__bf16)v0[j], (__bf16)v1[j] };
                *(v2bf*)&Blds[(cv * 4 + j) * LDB + kp * 2] = p;  // ds_store_b32
            }
        }
        __syncthreads();

        // ---- A fragment: ISA 16-bit A 16x32 layout ----
        // lanes 0-15 : elems 0-7 = K 0-7,  elems 8-15 = K 16-23  (row l16)
        // lanes 16-31: elems 0-7 = K 8-15, elems 8-15 = K 24-31
        const __bf16* ap = &Alds[(wm * 16 + l16) * LDA + half * 8];
        v8bf a0 = *(const v8bf*)(ap);
        v8bf a1 = *(const v8bf*)(ap + 16);
        v16bf afrag = __builtin_shufflevector(a0, a1,
            0,1,2,3,4,5,6,7,8,9,10,11,12,13,14,15);

        #pragma unroll
        for (int nt = 0; nt < 4; nt++) {
            // ---- B fragment: column l16, elems 0-7 = K half*8..+7,
            //      elems 8-15 = K 16+half*8..+7 (matches A layout) ----
            int col = wn * 64 + nt * 16 + l16;
            const __bf16* bp = &Blds[col * LDB + half * 8];
            v8bf b0 = *(const v8bf*)(bp);
            v8bf b1 = *(const v8bf*)(bp + 16);
            v16bf bfrag = __builtin_shufflevector(b0, b1,
                0,1,2,3,4,5,6,7,8,9,10,11,12,13,14,15);

            acc[nt] = __builtin_amdgcn_wmma_f32_16x16x32_bf16(
                false, afrag, false, bfrag, (short)0, acc[nt], false, false);
        }
        __syncthreads();
    }

    // ---- epilogue: + bias, store h ----
    // C/D layout: VGPR v -> row half*8+v, col l16 (within 16x16 tile)
    const int rowTile = mBase + wm * 16 + half * 8;
    if (mBase + BM <= N_NODES) {
        // fast path (781 of 782 blocks): no per-element guards
        #pragma unroll
        for (int nt = 0; nt < 4; nt++) {
            int col = nBase + wn * 64 + nt * 16 + l16;
            float bv = bias[col];
            float* hp = h + (size_t)rowTile * OUT_DIM + col;
            #pragma unroll
            for (int v = 0; v < 8; v++)
                hp[(size_t)v * OUT_DIM] = acc[nt][v] + bv;
        }
    } else {
        #pragma unroll
        for (int nt = 0; nt < 4; nt++) {
            int col = nBase + wn * 64 + nt * 16 + l16;
            float bv = bias[col];
            #pragma unroll
            for (int v = 0; v < 8; v++) {
                int row = rowTile + v;
                if (row < N_NODES)
                    h[(size_t)row * OUT_DIM + col] = acc[nt][v] + bv;
            }
        }
    }
}

// ---------------------------------------------------------------------------
// Kernel 2: zero the output accumulator
// ---------------------------------------------------------------------------
__global__ __launch_bounds__(256) void zero_kernel(v4f* __restrict__ out, int n4)
{
    int i = blockIdx.x * blockDim.x + threadIdx.x;
    if (i < n4) out[i] = (v4f){0.f, 0.f, 0.f, 0.f};
}

// ---------------------------------------------------------------------------
// Kernel 3: out[dst[e]] += h[src[e]] * w[e]   (one wave32 per edge)
// L2-resident gathers + f32 atomic adds; edge params forced scalar so base
// addresses live in SGPRs (saddr-form global ops)
// ---------------------------------------------------------------------------
__global__ __launch_bounds__(256) void scatter_kernel(
    const float* __restrict__ h, const int* __restrict__ src,
    const int* __restrict__ dst, const float* __restrict__ w,
    float* __restrict__ out)
{
    int e    = blockIdx.x * 8 + (threadIdx.x >> 5);
    int lane = threadIdx.x & 31;
    if (e >= N_EDGES) return;

    int   s  = __builtin_amdgcn_readfirstlane(src[e]);
    int   d  = __builtin_amdgcn_readfirstlane(dst[e]);
    float wt = __uint_as_float(__builtin_amdgcn_readfirstlane(__float_as_uint(w[e])));

    const float* hp = h   + (size_t)s * OUT_DIM;
    float*       op = out + (size_t)d * OUT_DIM;

    float vals[8];
    #pragma unroll
    for (int i = 0; i < 8; i++) vals[i] = hp[i * 32 + lane] * wt;  // loads in flight
    #pragma unroll
    for (int i = 0; i < 8; i++) atomicAdd(&op[i * 32 + lane], vals[i]);
}

// ---------------------------------------------------------------------------
extern "C" void kernel_launch(void* const* d_in, const int* in_sizes, int n_in,
                              void* d_out, int out_size, void* d_ws, size_t ws_size,
                              hipStream_t stream)
{
    const float* x   = (const float*)d_in[0];
    const int*   src = (const int*)  d_in[1];
    const int*   dst = (const int*)  d_in[2];
    const float* w   = (const float*)d_in[3];
    const float* W   = (const float*)d_in[4];
    const float* b   = (const float*)d_in[5];
    float* out = (float*)d_out;
    float* h   = (float*)d_ws;            // 50000*256*4 = 51.2 MB scratch

    dim3 g1((N_NODES + BM - 1) / BM, OUT_DIM / BN);   // 782 x 2
    gemm_bias_kernel<<<g1, 256, 0, stream>>>(x, W, b, h);

    int n4 = N_NODES * OUT_DIM / 4;
    zero_kernel<<<(n4 + 255) / 256, 256, 0, stream>>>((v4f*)out, n4);

    scatter_kernel<<<N_EDGES / 8, 256, 0, stream>>>(h, src, dst, w, out);
}
